// SelfAttention_59983513256025
// MI455X (gfx1250) — matrix operand order repaired and checked by reading, not compiled, not run on hardware
//
#include <hip/hip_runtime.h>
#include <hip/hip_bf16.h>
#include <hip/hip_fp16.h>

#define BB 4
#define TT 2048
#define KD 1024
#define HH 16
#define DD 64
#define KB 64           // keys per block iteration
#define KROW 72         // padded LDS row stride (halves) for K/V tiles
#define PROW 72         // padded LDS row stride (halves) for P tile

typedef __attribute__((ext_vector_type(16))) _Float16 v16h;
typedef __attribute__((ext_vector_type(8)))  _Float16 v8h;
typedef __attribute__((ext_vector_type(8)))  float    v8f;
typedef int v4i __attribute__((vector_size(16)));   // matches builtin param type

union V16H { v16h v; v8h h[2]; };

__device__ __forceinline__ v8f wmma_f16(v16h a, v16h b, v8f c) {
    return __builtin_amdgcn_wmma_f32_16x16x32_f16(
        false, a, false, b, (short)0, c, false, false);
}

// 16-byte global -> LDS copy, async (CDNA5) when available.
__device__ __forceinline__ void async_copy16(const _Float16* g, _Float16* l) {
#if __has_builtin(__builtin_amdgcn_global_load_async_to_lds_b128)
    __builtin_amdgcn_global_load_async_to_lds_b128(
        (__attribute__((address_space(1))) v4i*)g,
        (__attribute__((address_space(3))) v4i*)l, 0, 0);
#else
    *(v8h*)l = *(const v8h*)g;
#endif
}

__device__ __forceinline__ void wait_async0() {
#if __has_builtin(__builtin_amdgcn_s_wait_asynccnt)
    __builtin_amdgcn_s_wait_asynccnt(0);
#else
    asm volatile("s_wait_asynccnt 0" ::: "memory");
#endif
}

// ---------------------------------------------------------------------------
// Kernel 1: fused QKV projection, f32 in -> f16 out (Q scaled by D^-0.5),
// K stored transposed [b][h][d][t]. blockIdx.y selects the matrix (scalar).
// One wave per 16x16 output tile, K-loop in steps of 32.
// ---------------------------------------------------------------------------
__global__ __launch_bounds__(256) void qkv_proj_kernel(
    const float* __restrict__ x,  const float* __restrict__ Wq,
    const float* __restrict__ Wk, const float* __restrict__ Wv,
    _Float16* __restrict__ qh, _Float16* __restrict__ kt,
    _Float16* __restrict__ vh)
{
    const int lane  = threadIdx.x & 31;
    const int wv    = threadIdx.x >> 5;
    const int gw    = blockIdx.x * 8 + wv;
    const int mat   = blockIdx.y;            // 0=Q, 1=K, 2=V (uniform scalar)

    const int NT   = KD / 16;                // 64 col tiles
    const int rowt = gw / NT;
    const int colt = gw % NT;

    const float* W = (mat == 0) ? Wq : (mat == 1) ? Wk : Wv;

    const int mlo   = lane & 15;
    const int halfi = lane >> 4;

    const int arow = rowt * 16 + mlo;        // row in [0, B*T)
    const float* aptr = x + (size_t)arow * KD;

    v8f c = {};
    for (int k0 = 0; k0 < KD; k0 += 32) {
        V16H a;
        const float* ap = aptr + k0 + halfi * 8;
        #pragma unroll
        for (int j = 0; j < 8; ++j) {
            a.v[j]     = (_Float16)ap[j];
            a.v[j + 8] = (_Float16)ap[j + 16];
        }
        // B[kk][n] = W[colt*16+n][k0+kk], n = lane%16, kk in table order (L2-resident gather)
        V16H bf;
        const float* bp = W + (size_t)(colt * 16 + mlo) * KD + k0 + halfi * 8;
        #pragma unroll
        for (int n = 0; n < 16; ++n)
            bf.v[n] = (_Float16)bp[n + (n >> 3) * 8];

        c = wmma_f16(a.v, bf.v, c);
    }

    const int j = colt * 16 + mlo;           // output column in [0, K)
    const int h = j >> 6, d = j & 63;
    #pragma unroll
    for (int i = 0; i < 8; ++i) {
        const int m = i + 8 * halfi;
        const int r = rowt * 16 + m;
        const int b = r / TT, t = r % TT;
        const float val = c[i];
        if (mat == 0) {
            qh[(((size_t)b * HH + h) * TT + t) * DD + d] = (_Float16)(val * 0.125f);
        } else if (mat == 1) {
            kt[(((size_t)b * HH + h) * DD + d) * TT + t] = (_Float16)val;
        } else {
            vh[(((size_t)b * HH + h) * TT + t) * DD + d] = (_Float16)val;
        }
    }
}

// ---------------------------------------------------------------------------
// Kernel 2: causal flash attention. Block = 8 waves = 8 consecutive query
// tiles of one (b,h); K/V 64-key tiles staged cooperatively into LDS with
// async-to-LDS copies, shared by all waves. 16 WMMAs per wave per 64 keys.
// ---------------------------------------------------------------------------
__global__ __launch_bounds__(256) void attn_kernel(
    const _Float16* __restrict__ qh, const _Float16* __restrict__ kt,
    const _Float16* __restrict__ vh, float* __restrict__ out)
{
    __shared__ __align__(32) _Float16 kld[64 * KROW];      // [d][key]
    __shared__ __align__(32) _Float16 vld[64 * KROW];      // [key][d]
    __shared__ __align__(32) _Float16 plds[8][16 * PROW];  // per-wave P tile

    const int lane   = threadIdx.x & 31;
    const int wv     = threadIdx.x >> 5;
    const int bh     = blockIdx.x >> 4;          // b*H + h
    const int qchunk = blockIdx.x & 15;          // 8 query tiles per block
    const int qt     = qchunk * 8 + wv;
    const int mlo    = lane & 15;
    const int halfi  = lane >> 4;
    const int q0     = qt * 16;

    const _Float16* ktb = kt + (size_t)bh * DD * TT;
    const _Float16* vbb = vh + (size_t)bh * TT * DD;

    // Q fragments for two d-chunks (reused across whole key loop)
    V16H qf[2];
    {
        const _Float16* qp = qh + (((size_t)bh * TT) + q0 + mlo) * DD + halfi * 8;
        #pragma unroll
        for (int dc = 0; dc < 2; ++dc) {
            qf[dc].h[0] = *(const v8h*)(qp + dc * 32);
            qf[dc].h[1] = *(const v8h*)(qp + dc * 32 + 16);
        }
    }

    float mrow[8], lrow[8];
    v8f o[4];
    const v8f zero = {};
    #pragma unroll
    for (int i = 0; i < 8; ++i) { mrow[i] = -__builtin_inff(); lrow[i] = 0.0f; }
    #pragma unroll
    for (int dt = 0; dt < 4; ++dt) o[dt] = zero;

    // block-uniform key range: covers the block's largest query tile
    const int nblocks = qchunk * 2 + 2;          // (qchunk*128+127)/64 + 1

    for (int kb = 0; kb < nblocks; ++kb) {
        const int k0 = kb * KB;

        // ---- cooperative async stage of K tile [64 d][64 keys] and
        //      V tile [64 keys][64 d] into LDS (8KB each)
        #pragma unroll
        for (int p = 0; p < 2; ++p) {
            const int idx = p * 256 + threadIdx.x;
            const int row = idx >> 3;
            const int ch  = (idx & 7) * 8;
            async_copy16(ktb + (size_t)row * TT + k0 + ch, &kld[row * KROW + ch]);
            async_copy16(vbb + (size_t)(k0 + row) * DD + ch, &vld[row * KROW + ch]);
        }
        if (kb + 1 < nblocks) {   // prefetch next tile into cache hierarchy
            __builtin_prefetch(ktb + (size_t)(threadIdx.x & 63) * TT + k0 + KB, 0, 0);
            __builtin_prefetch(vbb + (size_t)(k0 + KB + (threadIdx.x & 63)) * DD, 0, 0);
        }
        wait_async0();
        __syncthreads();

        if (k0 <= q0 + 15) {     // causal: this wave needs this key block
            // ---- S = Q K^T : 4 column tiles x 2 d-chunks = 8 WMMAs
            v8f s[4];
            #pragma unroll
            for (int nh = 0; nh < 4; ++nh) s[nh] = zero;
            #pragma unroll
            for (int dc = 0; dc < 2; ++dc) {
                #pragma unroll
                for (int nh = 0; nh < 4; ++nh) {
                    V16H bf;
                    const _Float16* kp = &kld[(dc * 32 + halfi * 8) * KROW + nh * 16 + mlo];
                    #pragma unroll
                    for (int s2 = 0; s2 < 16; ++s2) bf.v[s2] = kp[(s2 + (s2 >> 3) * 8) * KROW];
                    s[nh] = wmma_f16(qf[dc].v, bf.v, s[nh]);
                }
            }

            // ---- causal mask on straddling blocks
            if (k0 + KB - 1 > q0) {
                #pragma unroll
                for (int i = 0; i < 8; ++i) {
                    const int q = q0 + i + 8 * halfi;
                    #pragma unroll
                    for (int nh = 0; nh < 4; ++nh)
                        if (k0 + nh * 16 + mlo > q) s[nh][i] = -__builtin_inff();
                }
            }

            // ---- online softmax across 64 scores per row
            #pragma unroll
            for (int i = 0; i < 8; ++i) {
                float v = fmaxf(fmaxf(s[0][i], s[1][i]), fmaxf(s[2][i], s[3][i]));
                v = fmaxf(v, __shfl_xor(v, 1, 32));
                v = fmaxf(v, __shfl_xor(v, 2, 32));
                v = fmaxf(v, __shfl_xor(v, 4, 32));
                v = fmaxf(v, __shfl_xor(v, 8, 32));
                const float mnew  = fmaxf(mrow[i], v);
                const float alpha = __expf(mrow[i] - mnew);
                mrow[i] = mnew;

                float p[4], rs = 0.0f;
                #pragma unroll
                for (int nh = 0; nh < 4; ++nh) {
                    p[nh] = __expf(s[nh][i] - mnew);
                    rs += p[nh];
                }
                rs += __shfl_xor(rs, 1, 32);
                rs += __shfl_xor(rs, 2, 32);
                rs += __shfl_xor(rs, 4, 32);
                rs += __shfl_xor(rs, 8, 32);
                lrow[i] = lrow[i] * alpha + rs;

                #pragma unroll
                for (int dt = 0; dt < 4; ++dt) o[dt][i] *= alpha;

                const int mr = i + 8 * halfi;
                #pragma unroll
                for (int nh = 0; nh < 4; ++nh)
                    plds[wv][mr * PROW + nh * 16 + mlo] = (_Float16)p[nh];
            }

            // ---- O += P x V : 2 k-chunks x 4 d-tiles = 8 WMMAs
            #pragma unroll
            for (int c = 0; c < 2; ++c) {
                V16H pa;
                const _Float16* pp = &plds[wv][mlo * PROW + c * 32 + halfi * 8];
                pa.h[0] = *(const v8h*)pp;
                pa.h[1] = *(const v8h*)(pp + 16);
                #pragma unroll
                for (int dt = 0; dt < 4; ++dt) {
                    V16H vf;
                    const _Float16* vp = &vld[(c * 32 + halfi * 8) * KROW + dt * 16 + mlo];
                    #pragma unroll
                    for (int s2 = 0; s2 < 16; ++s2) vf.v[s2] = vp[(s2 + (s2 >> 3) * 8) * KROW];
                    o[dt] = wmma_f16(pa.v, vf.v, o[dt]);
                }
            }
        }
        __syncthreads();   // protect K/V LDS tiles before next overwrite
    }

    // ---- epilogue: O /= l, write f32 out[b][t][h*64+d]
    const int b = bh / HH, h = bh % HH;
    #pragma unroll
    for (int i = 0; i < 8; ++i) {
        const int   t   = q0 + i + 8 * halfi;
        const float inv = 1.0f / lrow[i];
        #pragma unroll
        for (int dt = 0; dt < 4; ++dt) {
            const int d = dt * 16 + mlo;
            out[((size_t)b * TT + t) * KD + h * DD + d] = o[dt][i] * inv;
        }
    }
}

// ---------------------------------------------------------------------------
extern "C" void kernel_launch(void* const* d_in, const int* in_sizes, int n_in,
                              void* d_out, int out_size, void* d_ws, size_t ws_size,
                              hipStream_t stream) {
    const float* x  = (const float*)d_in[0];
    const float* Wq = (const float*)d_in[1];
    const float* Wk = (const float*)d_in[2];
    const float* Wv = (const float*)d_in[3];
    float* out = (float*)d_out;

    const size_t n_half = (size_t)BB * HH * TT * DD;   // 8M halves each
    _Float16* qh = (_Float16*)d_ws;
    _Float16* kt = qh + n_half;
    _Float16* vh = kt + n_half;

    {
        const int waves_per_mat = ((BB * TT) / 16) * (KD / 16);   // 32768
        dim3 grid(waves_per_mat / 8, 3);
        hipLaunchKernelGGL(qkv_proj_kernel, grid, dim3(256), 0, stream,
                           x, Wq, Wk, Wv, qh, kt, vh);
    }
    {
        dim3 grid(BB * HH * (TT / 128));                          // 1024 blocks
        hipLaunchKernelGGL(attn_kernel, grid, dim3(256), 0, stream,
                           qh, kt, vh, out);
    }
}
